// AttentionBlock_317827580342
// MI455X (gfx1250) — matrix operand order repaired and checked by reading, not compiled, not run on hardware
//
#include <hip/hip_runtime.h>
#include <hip/hip_bf16.h>

typedef __attribute__((ext_vector_type(16))) _Float16 v16h;
typedef __attribute__((ext_vector_type(8)))  float    v8f;

#define N_NODES 20000
#define N_PAIRS 320000
#define FDIM    192
#define KRBF    32
#define NORD    15

// ---------------------------------------------------------------------------
// LDS staging per wave (16 pairs). A-matrix layout tiles are 32B/lane so the
// WMMA A reads are two ds_load_b128 per k-step.
// ---------------------------------------------------------------------------
struct __align__(32) WaveShared {
    _Float16 aRad[6][32][16];   // hidden activations (K=192) in WMMA A layout
    _Float16 aSph[2][32][16];   // spherical hidden (K=48 padded to 64) A layout
    float dev[16][3];
    float cutS[16];
    int   iS[16];
    int   jS[16];
    float alpha1[16][4];
    float alpha2[16][3];
};

// fast silu: x * rcp(1 + exp(-x)) using the native v_rcp_f32 (no IEEE-div
// Newton fixup chains).
__device__ __forceinline__ float silu_f(float x) {
    return x * __builtin_amdgcn_rcpf(1.0f + __expf(-x));
}

// map global hidden index k32 (within a 32-wide k-step) to (laneHalf, halfIdx)
__device__ __forceinline__ void a_layout(int k32, int& halfsel, int& j) {
    halfsel = ((k32 & 15) >= 8) ? 1 : 0;
    j = (k32 < 16) ? (k32 - 8 * halfsel) : (8 + (k32 - 16 - 8 * halfsel));
}

// ---------------------------------------------------------------------------
// Zero the (poisoned) output buffer.
// ---------------------------------------------------------------------------
__global__ void zero_out_kernel(float* out, int n) {
    for (int i = blockIdx.x * blockDim.x + threadIdx.x; i < n;
         i += gridDim.x * blockDim.x)
        out[i] = 0.0f;
}

// ---------------------------------------------------------------------------
// Convert f32 weights [Ktot x 192] into f16 WMMA-B tiles.
// Tile (nt, ks): 32 lanes x 16 halves; lane l: n = nt*16+(l&15),
// kbase = ks*32 + (l>>4)*8, half j -> K = kbase + j (j<8) or kbase + 8 + j (j>=8). Rows K>=Ktot -> 0.
// ---------------------------------------------------------------------------
__global__ void prep_weights_kernel(const float* __restrict__ src,
                                    _Float16* __restrict__ dst,
                                    int Ktot, int KS) {
    int t = blockIdx.x * blockDim.x + threadIdx.x;
    int total = 12 * KS * 32;
    if (t >= total) return;
    int lane = t & 31;
    int tile = t >> 5;                 // nt*KS + ks
    int ks = tile % KS;
    int nt = tile / KS;
    int n = nt * 16 + (lane & 15);
    int kbase = ks * 32 + ((lane >> 4) * 8);
    _Float16* out = dst + (size_t)tile * 512 + (size_t)lane * 16;
#pragma unroll
    for (int j = 0; j < 16; ++j) {
        int K = kbase + ((j < 8) ? j : (8 + j));
        out[j] = (K < Ktot) ? (_Float16)src[(size_t)K * FDIM + n] : (_Float16)0.0f;
    }
}

// ---------------------------------------------------------------------------
// Node projections: q1,k1,v1 (per-head 48x48), q2,k2 (per-degree 64x64).
// blockIdx.y selects the array. silu on q/k, not on v1.
// ---------------------------------------------------------------------------
__global__ void proj_kernel(const float* __restrict__ x,
                            const float* __restrict__ Wq1,
                            const float* __restrict__ Wk1,
                            const float* __restrict__ Wv1,
                            const float* __restrict__ Wq2,
                            const float* __restrict__ Wk2,
                            float* __restrict__ q1, float* __restrict__ k1,
                            float* __restrict__ v1, float* __restrict__ q2,
                            float* __restrict__ k2) {
    int t = blockIdx.x * blockDim.x + threadIdx.x;
    if (t >= N_NODES * FDIM) return;
    int n = t / FDIM;
    int f = t - n * FDIM;
    int arr = blockIdx.y;
    const float* xr = x + (size_t)n * FDIM;
    float acc;
    if (arr < 3) {
        int h = f / 48, i = f - h * 48;
        const float* W = (arr == 0 ? Wq1 : (arr == 1 ? Wk1 : Wv1))
                         + ((size_t)(h * 48 + i)) * 48;
        const float* xv = xr + h * 48;
        acc = 0.0f;
#pragma unroll 8
        for (int jj = 0; jj < 48; ++jj) acc += W[jj] * xv[jj];
        float* out = (arr == 0 ? q1 : (arr == 1 ? k1 : v1));
        out[(size_t)n * FDIM + f] = (arr == 2) ? acc : silu_f(acc);
    } else {
        int l = f / 64, i = f - l * 64;
        const float* W = (arr == 3 ? Wq2 : Wk2) + ((size_t)(l * 64 + i)) * 64;
        const float* xv = xr + l * 64;
        acc = 0.0f;
#pragma unroll 8
        for (int jj = 0; jj < 64; ++jj) acc += W[jj] * xv[jj];
        float* out = (arr == 3 ? q2 : k2);
        out[(size_t)n * FDIM + f] = silu_f(acc);
    }
}

// ---------------------------------------------------------------------------
// One filter phase: layer1 WMMA (K=32) -> silu -> LDS (A layout), spherical
// layer1 (3->48) -> LDS, layer2 WMMA (radial K=192 + spherical K=64 padded)
// with TWO independent accumulators to break the D->C dependency chain, then
// attention-logit reduction straight from the C layout (FH=48 = 3 tiles/head,
// FL=64 = 4 tiles/degree).
// ---------------------------------------------------------------------------
__device__ __forceinline__ void run_phase(
    WaveShared& S, int lane, const v16h& aR,
    const _Float16* __restrict__ raT, const float* __restrict__ ra_b,
    const _Float16* __restrict__ rbT, const float* __restrict__ rb_b,
    const float* __restrict__ sa_w, const float* __restrict__ sa_b,
    const _Float16* __restrict__ sbT, const float* __restrict__ sb_b,
    const float* __restrict__ qarr, const float* __restrict__ karr,
    int grp, float scale, float* alphaOut, int apitch) {
    const int col = lane & 15;
    const int mhi = (lane >> 4) * 8;

    // ---- layer 1 (radial): 12 x wmma 16x16x32; unroll 2 so the scheduler
    // can overlap silu VALU of tile n with the wmma of tile n+1 ----
#pragma unroll 2
    for (int nt = 0; nt < 12; ++nt) {
        v8f c = {};
        v16h b = *(const v16h*)(raT + ((size_t)nt * 32 + lane) * 16);
        c = __builtin_amdgcn_wmma_f32_16x16x32_f16(false, aR, false, b,
                                                   (short)0, c, false, false);
        int n = nt * 16 + col;
        float bias = ra_b[n];
        int ks = n >> 5, k32 = n & 31, halfsel, j;
        a_layout(k32, halfsel, j);
#pragma unroll
        for (int r = 0; r < 8; ++r) {
            S.aRad[ks][r + mhi + 16 * halfsel][j] =
                (_Float16)silu_f(c[r] + bias);
        }
    }

    // ---- spherical layer 1: dev(3) -> 48 hidden ----
    for (int it = 0; it < 2; ++it) {
        int u = lane + 32 * it;
        if (u < 48) {
            float w0 = sa_w[u], w1 = sa_w[48 + u], w2 = sa_w[96 + u];
            float bb = sa_b[u];
            int ks = u >> 5, k32 = u & 31, halfsel, j;
            a_layout(k32, halfsel, j);
#pragma unroll
            for (int p = 0; p < 16; ++p) {
                float v = bb + S.dev[p][0] * w0 + S.dev[p][1] * w1 +
                          S.dev[p][2] * w2;
                S.aSph[ks][p + 16 * halfsel][j] = (_Float16)silu_f(v);
            }
        }
    }
    __syncthreads();

    // ---- layer 2 (radial + spherical fused) + alpha partials ----
    float part[8];
#pragma unroll
    for (int r = 0; r < 8; ++r) part[r] = 0.0f;

    for (int nt = 0; nt < 12; ++nt) {
        v8f c0 = {};
        v8f c1 = {};
#pragma unroll
        for (int ks = 0; ks < 6; ks += 2) {
            v16h a0 = *(const v16h*)&S.aRad[ks][lane][0];
            v16h b0 = *(const v16h*)(rbT + ((size_t)(nt * 6 + ks) * 32 + lane) * 16);
            c0 = __builtin_amdgcn_wmma_f32_16x16x32_f16(false, a0, false, b0,
                                                        (short)0, c0, false, false);
            v16h a1 = *(const v16h*)&S.aRad[ks + 1][lane][0];
            v16h b1 = *(const v16h*)(rbT + ((size_t)(nt * 6 + ks + 1) * 32 + lane) * 16);
            c1 = __builtin_amdgcn_wmma_f32_16x16x32_f16(false, a1, false, b1,
                                                        (short)0, c1, false, false);
        }
        {
            v16h a0 = *(const v16h*)&S.aSph[0][lane][0];
            v16h b0 = *(const v16h*)(sbT + ((size_t)(nt * 2) * 32 + lane) * 16);
            c0 = __builtin_amdgcn_wmma_f32_16x16x32_f16(false, a0, false, b0,
                                                        (short)0, c0, false, false);
            v16h a1 = *(const v16h*)&S.aSph[1][lane][0];
            v16h b1 = *(const v16h*)(sbT + ((size_t)(nt * 2 + 1) * 32 + lane) * 16);
            c1 = __builtin_amdgcn_wmma_f32_16x16x32_f16(false, a1, false, b1,
                                                        (short)0, c1, false, false);
        }
        int n = nt * 16 + col;
        float bias = rb_b[n] + sb_b[n];
#pragma unroll
        for (int r = 0; r < 8; ++r) {
            int M = r + mhi;
            float w = c0[r] + c1[r] + bias;
            part[r] += w * qarr[(size_t)S.iS[M] * FDIM + n] *
                           karr[(size_t)S.jS[M] * FDIM + n];
        }
        if ((nt % grp) == grp - 1) {
            // reduce over the 16 column-lanes of each half-wave
            for (int m = 1; m < 16; m <<= 1)
#pragma unroll
                for (int r = 0; r < 8; ++r)
                    part[r] += __shfl_xor(part[r], m, 32);
            if (col == 0) {
                int g = nt / grp;
#pragma unroll
                for (int r = 0; r < 8; ++r) {
                    int M = r + mhi;
                    alphaOut[M * apitch + g] = part[r] * S.cutS[M] * scale;
                }
            }
#pragma unroll
            for (int r = 0; r < 8; ++r) part[r] = 0.0f;
        }
    }
    __syncthreads();
}

// ---------------------------------------------------------------------------
// Fused per-pair kernel: 4 waves/block, 16 pairs/wave, 64 pairs/block.
// ---------------------------------------------------------------------------
__global__ __launch_bounds__(128) void pair_kernel(
    const float* __restrict__ rbf, const float* __restrict__ ev,
    const float* __restrict__ ylm, const float* __restrict__ cut,
    const int* __restrict__ idx_i, const int* __restrict__ idx_j,
    const float* __restrict__ r1a_b, const float* __restrict__ r1b_b,
    const float* __restrict__ r2a_b, const float* __restrict__ r2b_b,
    const float* __restrict__ s1a_w, const float* __restrict__ s1a_b,
    const float* __restrict__ s1b_b, const float* __restrict__ s2a_w,
    const float* __restrict__ s2a_b, const float* __restrict__ s2b_b,
    const _Float16* __restrict__ r1aT, const _Float16* __restrict__ r1bT,
    const _Float16* __restrict__ r2aT, const _Float16* __restrict__ r2bT,
    const _Float16* __restrict__ s1bT, const _Float16* __restrict__ s2bT,
    const float* __restrict__ q1, const float* __restrict__ k1,
    const float* __restrict__ v1, const float* __restrict__ q2,
    const float* __restrict__ k2, float* __restrict__ out_x,
    float* __restrict__ out_ev) {
    __shared__ WaveShared smem[4];
    const int lane = threadIdx.x & 31;
    const int wid = threadIdx.x >> 5;
    WaveShared& S = smem[wid];
    const int base = (blockIdx.x * 4 + wid) * 16;

    // per-pair scalars
    if (lane < 16) {
        S.iS[lane] = idx_i[base + lane];
        S.jS[lane] = idx_j[base + lane];
        S.cutS[lane] = cut[base + lane];
    }

    // A-matrix for rbf (16 pairs x K=32), f32 -> f16, WMMA A layout
    v16h aR;
    {
        int row = base + (lane & 15);
        int koff = (lane >> 4) * 8;
        const float* rp = rbf + (size_t)row * KRBF;
#pragma unroll
        for (int j = 0; j < 16; ++j) {
            int K = (j < 8) ? (koff + j) : (16 + koff + (j - 8));
            aR[j] = (_Float16)rp[K];
        }
    }
    __syncthreads();

    // dev = per-degree sum of squares of ev[j]-ev[i]
    if (lane < 16) {
        const float* evj = ev + (size_t)S.jS[lane] * NORD;
        const float* evi = ev + (size_t)S.iS[lane] * NORD;
        float d0 = 0.f, d1 = 0.f, d2 = 0.f;
#pragma unroll
        for (int o = 0; o < NORD; ++o) {
            float d = evj[o] - evi[o];
            float dd = d * d;
            if (o < 3) d0 += dd; else if (o < 8) d1 += dd; else d2 += dd;
        }
        S.dev[lane][0] = d0; S.dev[lane][1] = d1; S.dev[lane][2] = d2;
    }
    // zero the K=48..63 padding of the spherical A tile (once; both phases)
    {
        _Float16* z = &S.aSph[1][lane][8];
#pragma unroll
        for (int j = 0; j < 8; ++j) z[j] = (_Float16)0.0f;
    }
    __syncthreads();

    // ---------------- phase 1: w1 / alpha1 / x_att ----------------
    run_phase(S, lane, aR, r1aT, r1a_b, r1bT, r1b_b, s1a_w, s1a_b, s1bT, s1b_b,
              q1, k1, /*grp=*/3, /*1/sqrt(48)=*/0.14433756729740643f,
              &S.alpha1[0][0], 4);

    // x_att: idx_i is sorted, so run-length-aggregate contributions that
    // target the same node before issuing global atomics.
    for (int it = 0; it < 6; ++it) {
        int f = lane + 32 * it;
        int h = f / 48;
        int icur = S.iS[0];
        float acc = 0.0f;
        for (int pl = 0; pl < 16; ++pl) {
            int i = S.iS[pl];
            if (i != icur) {
                atomicAdd(&out_x[(size_t)icur * FDIM + f], acc);
                acc = 0.0f;
                icur = i;
            }
            acc += S.alpha1[pl][h] * v1[(size_t)S.jS[pl] * FDIM + f];
        }
        atomicAdd(&out_x[(size_t)icur * FDIM + f], acc);
    }
    __syncthreads();

    // ---------------- phase 2: w2 / alpha2 / ev_att ----------------
    run_phase(S, lane, aR, r2aT, r2a_b, r2bT, r2b_b, s2a_w, s2a_b, s2bT, s2b_b,
              q2, k2, /*grp=*/4, /*1/sqrt(64)=*/0.125f, &S.alpha2[0][0], 3);

    if (lane < NORD) {
        int o = lane;
        int deg = (o < 3) ? 0 : ((o < 8) ? 1 : 2);
        int icur = S.iS[0];
        float acc = 0.0f;
        for (int pl = 0; pl < 16; ++pl) {
            int i = S.iS[pl];
            if (i != icur) {
                atomicAdd(&out_ev[(size_t)icur * NORD + o], acc);
                acc = 0.0f;
                icur = i;
            }
            acc += S.alpha2[pl][deg] * ylm[(size_t)(base + pl) * NORD + o];
        }
        atomicAdd(&out_ev[(size_t)icur * NORD + o], acc);
    }
}

// ---------------------------------------------------------------------------
extern "C" void kernel_launch(void* const* d_in, const int* in_sizes, int n_in,
                              void* d_out, int out_size, void* d_ws,
                              size_t ws_size, hipStream_t stream) {
    const float* x     = (const float*)d_in[0];
    const float* ev    = (const float*)d_in[1];
    const float* rbf   = (const float*)d_in[2];
    const float* ylm   = (const float*)d_in[3];
    const float* cutv  = (const float*)d_in[4];
    const int*   idx_i = (const int*)d_in[5];
    const int*   idx_j = (const int*)d_in[6];
    const float* r1a_w = (const float*)d_in[7];
    const float* r1a_b = (const float*)d_in[8];
    const float* r1b_w = (const float*)d_in[9];
    const float* r1b_b = (const float*)d_in[10];
    const float* r2a_w = (const float*)d_in[11];
    const float* r2a_b = (const float*)d_in[12];
    const float* r2b_w = (const float*)d_in[13];
    const float* r2b_b = (const float*)d_in[14];
    const float* s1a_w = (const float*)d_in[15];
    const float* s1a_b = (const float*)d_in[16];
    const float* s1b_w = (const float*)d_in[17];
    const float* s1b_b = (const float*)d_in[18];
    const float* s2a_w = (const float*)d_in[19];
    const float* s2a_b = (const float*)d_in[20];
    const float* s2b_w = (const float*)d_in[21];
    const float* s2b_b = (const float*)d_in[22];
    const float* Wq1   = (const float*)d_in[23];
    const float* Wk1   = (const float*)d_in[24];
    const float* Wq2   = (const float*)d_in[25];
    const float* Wk2   = (const float*)d_in[26];
    const float* Wv1   = (const float*)d_in[27];

    // workspace layout
    float* q1 = (float*)d_ws;
    float* k1 = q1 + (size_t)N_NODES * FDIM;
    float* v1 = k1 + (size_t)N_NODES * FDIM;
    float* q2 = v1 + (size_t)N_NODES * FDIM;
    float* k2 = q2 + (size_t)N_NODES * FDIM;
    _Float16* wh   = (_Float16*)(k2 + (size_t)N_NODES * FDIM);
    _Float16* r1aT = wh;                    // 12*1 tiles * 512 halves
    _Float16* r1bT = r1aT + 12 * 512;       // 12*6 tiles
    _Float16* r2aT = r1bT + 72 * 512;
    _Float16* r2bT = r2aT + 12 * 512;
    _Float16* s1bT = r2bT + 72 * 512;       // 12*2 tiles
    _Float16* s2bT = s1bT + 24 * 512;

    float* out_x  = (float*)d_out;
    float* out_ev = out_x + (size_t)N_NODES * FDIM;

    // 1) zero the (poisoned) output — atomics accumulate into it
    zero_out_kernel<<<2048, 256, 0, stream>>>((float*)d_out, out_size);

    // 2) pre-swizzle weights into f16 WMMA-B tiles (hot in L2 forever after)
    prep_weights_kernel<<<(12 * 1 * 32 + 255) / 256, 256, 0, stream>>>(r1a_w, r1aT, 32, 1);
    prep_weights_kernel<<<(12 * 6 * 32 + 255) / 256, 256, 0, stream>>>(r1b_w, r1bT, 192, 6);
    prep_weights_kernel<<<(12 * 1 * 32 + 255) / 256, 256, 0, stream>>>(r2a_w, r2aT, 32, 1);
    prep_weights_kernel<<<(12 * 6 * 32 + 255) / 256, 256, 0, stream>>>(r2b_w, r2bT, 192, 6);
    prep_weights_kernel<<<(12 * 2 * 32 + 255) / 256, 256, 0, stream>>>(s1b_w, s1bT, 48, 2);
    prep_weights_kernel<<<(12 * 2 * 32 + 255) / 256, 256, 0, stream>>>(s2b_w, s2bT, 48, 2);

    // 3) node projections (q1,k1,v1,q2,k2) with fused silu
    proj_kernel<<<dim3((N_NODES * FDIM + 255) / 256, 5), 256, 0, stream>>>(
        x, Wq1, Wk1, Wv1, Wq2, Wk2, q1, k1, v1, q2, k2);

    // 4) fused per-pair WMMA kernel: 64 pairs/block
    pair_kernel<<<N_PAIRS / 64, 128, 0, stream>>>(
        rbf, ev, ylm, cutv, idx_i, idx_j, r1a_b, r1b_b, r2a_b, r2b_b, s1a_w,
        s1a_b, s1b_b, s2a_w, s2a_b, s2b_b, r1aT, r1bT, r2aT, r2bT, s1bT, s2bT,
        q1, k1, v1, q2, k2, out_x, out_ev);
}
